// Recommender_9414568312930
// MI455X (gfx1250) — compile-verified, matrix operand order checked by reading, not run on hardware
//
#include <hip/hip_runtime.h>
#include <math.h>

#define DIM 64
#define MIN_NORMF 1e-15f
#define ATANH_EPSF 1e-05f
#define BALL_EPSF 0.004f

typedef __attribute__((ext_vector_type(16))) _Float16 v16h;
typedef __attribute__((ext_vector_type(8)))  float    v8f;

__device__ __forceinline__ float waveSum(float v) {
#pragma unroll
    for (int o = 16; o > 0; o >>= 1) v += __shfl_xor(v, o, 32);
    return v;
}

// ---------------- init: src -> dst1 and dst2 (cur buffer + residual output) ----
__global__ void copy_dual_kernel(const float* __restrict__ src,
                                 float* __restrict__ dst1,
                                 float* __restrict__ dst2, size_t n) {
    size_t i = (size_t)blockIdx.x * blockDim.x + threadIdx.x;
    size_t stride = (size_t)gridDim.x * blockDim.x;
    for (; i < n; i += stride) { float v = src[i]; dst1[i] = v; dst2[i] = v; }
}

__global__ void zero_kernel(float* __restrict__ p, size_t n) {
    size_t i = (size_t)blockIdx.x * blockDim.x + threadIdx.x;
    size_t stride = (size_t)gridDim.x * blockDim.x;
    for (; i < n; i += stride) p[i] = 0.0f;
}

// ---------------- KG edge kernel: one wave32 per edge, 2 dims/lane ------------
__global__ void kg_edge_kernel(const float* __restrict__ ent,
                               const float* __restrict__ relw,
                               const int* __restrict__ edge_index,
                               const int* __restrict__ edge_type,
                               float* __restrict__ ent_acc,
                               float* __restrict__ cnt, int n_edges) {
    int wave = (int)((blockIdx.x * (size_t)blockDim.x + threadIdx.x) >> 5);
    int lane = threadIdx.x & 31;
    if (wave >= n_edges) return;
    int head = edge_index[wave];
    int tail = edge_index[n_edges + wave];
    int rel  = edge_type[wave] - 1;

    // contiguous pair per lane: dims {2*lane, 2*lane+1} -> single b64 load
    const float2* hp = (const float2*)(ent + (size_t)head * DIM) + lane;
    const float2* tp = (const float2*)(ent + (size_t)tail * DIM) + lane;
    const float2* rp = (const float2*)(relw + (size_t)rel * DIM) + lane;
    float2 hv = *hp, tv = *tp, rv = *rp;
    float h0 = hv.x, h1 = hv.y;
    float t0 = tv.x, t1 = tv.y;
    float r0 = rv.x, r1 = rv.y;

    // hh = expmap0(h)
    float h2 = waveSum(h0 * h0 + h1 * h1);
    float nh = fmaxf(sqrtf(h2), MIN_NORMF);
    float sh = tanhf(nh) / nh;
    float hh0 = sh * h0, hh1 = sh * h1;
    float hh2 = sh * sh * h2;
    float lam = 2.0f / fmaxf(1.0f - hh2, MIN_NORMF);

    // ht = expmap(t, hh) = mobius_add(hh, tanh(lam*|t|/2)*t/|t|)
    float t2 = waveSum(t0 * t0 + t1 * t1);
    float nt = fmaxf(sqrtf(t2), MIN_NORMF);
    float st = tanhf(lam * nt * 0.5f) / nt;
    float e0 = st * t0, e1 = st * t1;
    float y2 = st * st * t2;
    float xy = waveSum(hh0 * e0 + hh1 * e1);
    float cx = 1.0f + 2.0f * xy + y2;
    float cy = 1.0f - hh2;
    float dn = fmaxf(1.0f + 2.0f * xy + hh2 * y2, MIN_NORMF);
    float ht0 = (cx * hh0 + cy * e0) / dn;
    float ht1 = (cx * hh1 + cy * e1) / dn;

    // hr = expmap(r, hh)
    float r2 = waveSum(r0 * r0 + r1 * r1);
    float nr = fmaxf(sqrtf(r2), MIN_NORMF);
    float sr = tanhf(lam * nr * 0.5f) / nr;
    float f0 = sr * r0, f1 = sr * r1;
    float y2r = sr * sr * r2;
    float xyr = waveSum(hh0 * f0 + hh1 * f1);
    float cxr = 1.0f + 2.0f * xyr + y2r;
    float dnr = fmaxf(1.0f + 2.0f * xyr + hh2 * y2r, MIN_NORMF);
    float hr0 = (cxr * hh0 + cy * f0) / dnr;
    float hr1 = (cxr * hh1 + cy * f1) / dnr;

    // m = mobius_add(ht, hr)
    float x2m = waveSum(ht0 * ht0 + ht1 * ht1);
    float y2m = waveSum(hr0 * hr0 + hr1 * hr1);
    float xym = waveSum(ht0 * hr0 + ht1 * hr1);
    float cxm = 1.0f + 2.0f * xym + y2m;
    float cym = 1.0f - x2m;
    float dnm = fmaxf(1.0f + 2.0f * xym + x2m * y2m, MIN_NORMF);
    float m0 = (cxm * ht0 + cym * hr0) / dnm;
    float m1 = (cxm * ht1 + cym * hr1) / dnm;

    // p = project(m)
    float m2 = waveSum(m0 * m0 + m1 * m1);
    float nm = fmaxf(sqrtf(m2), MIN_NORMF);
    float maxn = 1.0f - BALL_EPSF;
    float sc = (nm > maxn) ? (maxn / nm) : 1.0f;
    float p0 = sc * m0, p1 = sc * m1;
    float p2s = sc * sc * m2;

    // res = logmap(p, hh): sub = mobius_add(-hh, p)
    float xys = waveSum(-hh0 * p0 - hh1 * p1);
    float cxs = 1.0f + 2.0f * xys + p2s;
    float cys = 1.0f - hh2;
    float dns = fmaxf(1.0f + 2.0f * xys + hh2 * p2s, MIN_NORMF);
    float s0 = (cxs * (-hh0) + cys * p0) / dns;
    float s1 = (cxs * (-hh1) + cys * p1) / dns;
    float s2 = waveSum(s0 * s0 + s1 * s1);
    float sn = fmaxf(sqrtf(s2), MIN_NORMF);
    float snc = fminf(fmaxf(sn, -1.0f + ATANH_EPSF), 1.0f - ATANH_EPSF);
    float coef = (2.0f / lam) * atanhf(snc) / sn;
    float res0 = coef * s0, res1 = coef * s1;

    float* dstp = ent_acc + (size_t)head * DIM + 2 * lane;
    unsafeAtomicAdd(dstp, res0);
    unsafeAtomicAdd(dstp + 1, res1);
    if (lane == 0) unsafeAtomicAdd(&cnt[head], 1.0f);
}

// ---------------- generic row scatter-add: dst[d] += w * src[s] ---------------
__global__ void scatter_rows_kernel(const float* __restrict__ src,
                                    const int* __restrict__ src_idx,
                                    const int* __restrict__ dst_idx,
                                    const float* __restrict__ val,
                                    float* __restrict__ dst, int n) {
    int wave = (int)((blockIdx.x * (size_t)blockDim.x + threadIdx.x) >> 5);
    int lane = threadIdx.x & 31;
    if (wave >= n) return;
    int s = src_idx[wave];
    int d = dst_idx[wave];
    float w = (val != nullptr) ? val[wave] : 1.0f;
    float2 v = *((const float2*)(src + (size_t)s * DIM) + lane);
    float* dstp = dst + (size_t)d * DIM + 2 * lane;
    unsafeAtomicAdd(dstp,     w * v.x);
    unsafeAtomicAdd(dstp + 1, w * v.y);
}

// ---------------- gate + fusion via WMMA f16 ----------------------------------
// gi = sigmoid(item_cf @ G1^T + item_kg @ G2^T); fusion = gi*cf + (1-gi)*kg
// One wave per 16x16 output tile; K=64 -> 2 x wmma per operand (4 wmma/wave).
// Fragment layout (16-bit A 16x32, ISA 7.12.2): lane half h holds two
// contiguous 8-element K runs: K = 8h..8h+7 and K = 16+8h..16+8h+7.
// -> pack each fragment from 4 x float4 (b128) loads, branch-free.
__device__ __forceinline__ v16h pack_frag(const float* __restrict__ base) {
    float4 a0 = *(const float4*)(base);
    float4 a1 = *(const float4*)(base + 4);
    float4 a2 = *(const float4*)(base + 16);
    float4 a3 = *(const float4*)(base + 20);
    v16h d;
    d[0]  = (_Float16)a0.x; d[1]  = (_Float16)a0.y;
    d[2]  = (_Float16)a0.z; d[3]  = (_Float16)a0.w;
    d[4]  = (_Float16)a1.x; d[5]  = (_Float16)a1.y;
    d[6]  = (_Float16)a1.z; d[7]  = (_Float16)a1.w;
    d[8]  = (_Float16)a2.x; d[9]  = (_Float16)a2.y;
    d[10] = (_Float16)a2.z; d[11] = (_Float16)a2.w;
    d[12] = (_Float16)a3.x; d[13] = (_Float16)a3.y;
    d[14] = (_Float16)a3.z; d[15] = (_Float16)a3.w;
    return d;
}

__global__ void gate_fusion_kernel(const float* __restrict__ item_cf,
                                   const float* __restrict__ ent,   // first n_items rows
                                   const float* __restrict__ g1,    // [64][64] row-major (n,k)
                                   const float* __restrict__ g2,
                                   float* __restrict__ fusion, int n_items) {
    int wave = (int)((blockIdx.x * (size_t)blockDim.x + threadIdx.x) >> 5);
    int lane = threadIdx.x & 31;
    int rowBlocks = (n_items + 15) >> 4;
    int nTiles = rowBlocks * 4;            // 64 output cols = 4 tiles of 16
    if (wave >= nTiles) return;            // whole wave exits: EXEC stays all-1 for WMMA
    int rb = wave >> 2;
    int n0 = (wave & 3) << 4;
    int m0 = rb << 4;
    int half = lane >> 4, lm = lane & 15;
    int mA = m0 + lm;                      // A-fragment row for this lane
    if (mA > n_items - 1) mA = n_items - 1; // branch-free clamp (stores stay guarded)
    int nB = n0 + lm;                      // B-fragment column for this lane

    v8f acc = {};
#pragma unroll
    for (int kp = 0; kp < 2; ++kp) {
        int koff = kp * 32 + 8 * half;     // start of this lane's first K run
        v16h a_cf = pack_frag(item_cf + (size_t)mA * DIM + koff);
        v16h a_kg = pack_frag(ent + (size_t)mA * DIM + koff);
        v16h b1   = pack_frag(g1 + (size_t)nB * DIM + koff);
        v16h b2   = pack_frag(g2 + (size_t)nB * DIM + koff);
        acc = __builtin_amdgcn_wmma_f32_16x16x32_f16(false, a_cf, false, b1,
                                                     (short)0, acc, false, false);
        acc = __builtin_amdgcn_wmma_f32_16x16x32_f16(false, a_kg, false, b2,
                                                     (short)0, acc, false, false);
    }

    // C/D layout: VGPR e -> M = m0 + e + 8*half, N = n0 + (lane&15)
#pragma unroll
    for (int e = 0; e < 8; ++e) {
        int m = m0 + e + 8 * half;
        if (m < n_items) {
            float g = 1.0f / (1.0f + __expf(-acc[e]));
            float cf = item_cf[(size_t)m * DIM + nB];
            float kg = ent[(size_t)m * DIM + nB];
            fusion[(size_t)m * DIM + nB] = g * cf + (1.0f - g) * kg;
        }
    }
}

// ---------------- finalize: mean (optional) + l2normalize + residual add ------
__global__ void finalize_rows_kernel(const float* __restrict__ acc,
                                     const float* __restrict__ cnt,
                                     float* __restrict__ cur,
                                     float* __restrict__ res, int n_rows) {
    int wave = (int)((blockIdx.x * (size_t)blockDim.x + threadIdx.x) >> 5);
    int lane = threadIdx.x & 31;
    if (wave >= n_rows) return;
    float2 a = *((const float2*)(acc + (size_t)wave * DIM) + lane);
    if (cnt != nullptr) {
        float c = fmaxf(cnt[wave], 1.0f);
        a.x /= c; a.y /= c;
    }
    float n2 = waveSum(a.x * a.x + a.y * a.y);
    float inv = 1.0f / fmaxf(sqrtf(n2), 1e-12f);
    float2 v; v.x = a.x * inv; v.y = a.y * inv;
    *((float2*)(cur + (size_t)wave * DIM) + lane) = v;
    float2* rp = (float2*)(res + (size_t)wave * DIM) + lane;
    float2 r = *rp;
    r.x += v.x; r.y += v.y;
    *rp = r;
}

// -----------------------------------------------------------------------------
extern "C" void kernel_launch(void* const* d_in, const int* in_sizes, int n_in,
                              void* d_out, int out_size, void* d_ws, size_t ws_size,
                              hipStream_t stream) {
    const float* user_emb   = (const float*)d_in[0];
    const float* entity_emb = (const float*)d_in[1];
    const float* item_emb   = (const float*)d_in[2];
    const float* relw       = (const float*)d_in[3];
    const float* gate1      = (const float*)d_in[4];
    const float* gate2      = (const float*)d_in[5];
    const int*   edge_index = (const int*)d_in[6];
    const int*   edge_type  = (const int*)d_in[7];
    const int*   mat_row    = (const int*)d_in[8];
    const int*   mat_col    = (const int*)d_in[9];
    const float* mat_val    = (const float*)d_in[10];

    const int n_users = in_sizes[0] / DIM;
    const int n_ents  = in_sizes[1] / DIM;
    const int n_items = in_sizes[2] / DIM;
    const int n_edges = in_sizes[6] / 2;
    const int nnz     = in_sizes[8];
    const int n_hops  = in_sizes[4] / (DIM * DIM);

    const size_t entN  = (size_t)n_ents * DIM;
    const size_t userN = (size_t)n_users * DIM;
    const size_t itemN = (size_t)n_items * DIM;

    float* ws       = (float*)d_ws;
    float* ent_cur  = ws;
    float* user_cur = ent_cur + entN;
    float* item_cur = user_cur + userN;
    float* ent_acc  = item_cur + itemN;   // start of contiguous zeroed region
    float* user_acc = ent_acc + entN;
    float* item_acc = user_acc + userN;
    float* cnt      = item_acc + itemN;
    float* fusion   = cnt + n_ents;
    const size_t accLen = entN + userN + itemN + (size_t)n_ents;

    float* ent_res  = (float*)d_out;
    float* user_res = ent_res + entN;
    float* item_res = user_res + userN;

    const int TPB = 256;
    const int WPB = TPB / 32;
    const int CPY_BLOCKS = 2048;

    // init residuals + working copies
    copy_dual_kernel<<<CPY_BLOCKS, TPB, 0, stream>>>(entity_emb, ent_cur, ent_res, entN);
    copy_dual_kernel<<<CPY_BLOCKS, TPB, 0, stream>>>(user_emb, user_cur, user_res, userN);
    copy_dual_kernel<<<CPY_BLOCKS, TPB, 0, stream>>>(item_emb, item_cur, item_res, itemN);

    for (int hop = 0; hop < n_hops; ++hop) {
        // zero all accumulators + counts in one pass
        zero_kernel<<<CPY_BLOCKS, TPB, 0, stream>>>(ent_acc, accLen);

        // (a) KG hyperbolic edge aggregation -> ent_acc, cnt
        kg_edge_kernel<<<(n_edges + WPB - 1) / WPB, TPB, 0, stream>>>(
            ent_cur, relw, edge_index, edge_type, ent_acc, cnt, n_edges);

        // (b) item_agg_cf = segment_sum(user_cur[mat_row]) over mat_col
        scatter_rows_kernel<<<(nnz + WPB - 1) / WPB, TPB, 0, stream>>>(
            user_cur, mat_row, mat_col, nullptr, item_acc, nnz);

        // (c) gate + fusion via WMMA
        {
            int rowBlocks = (n_items + 15) / 16;
            int nTiles = rowBlocks * 4;
            gate_fusion_kernel<<<(nTiles + WPB - 1) / WPB, TPB, 0, stream>>>(
                item_cur, ent_cur, gate1 + (size_t)hop * DIM * DIM,
                gate2 + (size_t)hop * DIM * DIM, fusion, n_items);
        }

        // (d) user_agg = segment_sum(mat_val * fusion[mat_col]) over mat_row
        scatter_rows_kernel<<<(nnz + WPB - 1) / WPB, TPB, 0, stream>>>(
            fusion, mat_col, mat_row, mat_val, user_acc, nnz);

        // (e) finalize: mean/l2norm + residual accumulate
        finalize_rows_kernel<<<(n_ents + WPB - 1) / WPB, TPB, 0, stream>>>(
            ent_acc, cnt, ent_cur, ent_res, n_ents);
        finalize_rows_kernel<<<(n_users + WPB - 1) / WPB, TPB, 0, stream>>>(
            user_acc, nullptr, user_cur, user_res, n_users);
        finalize_rows_kernel<<<(n_items + WPB - 1) / WPB, TPB, 0, stream>>>(
            item_acc, nullptr, item_cur, item_res, n_items);
    }
}
